// MultiHeadAttention_25512105738967
// MI455X (gfx1250) — matrix operand order repaired and checked by reading, not compiled, not run on hardware
//
#include <hip/hip_runtime.h>
#include <hip/hip_bf16.h>

typedef __attribute__((ext_vector_type(16))) _Float16 v16h;
typedef __attribute__((ext_vector_type(8)))  _Float16 v8h;
typedef __attribute__((ext_vector_type(8)))  float    v8f;

#define BATCH 2
#define SEQ   2048
#define DMOD  1024
#define NHEAD 16
#define HDIM  64
// softmax scale (1/sqrt(64)) folded with log2(e) so probs = exp2(scores - m)
#define QSCALE 0.18033688011112042f

// ---------------------------------------------------------------------------
// helpers
// ---------------------------------------------------------------------------
__device__ __forceinline__ v16h load16h(const _Float16* p0, const _Float16* p1) {
    v16h r;
    *(v8h*)&r           = *(const v8h*)p0;   // 16B load -> elements 0..7
    *(((v8h*)&r) + 1)   = *(const v8h*)p1;   // 16B load -> elements 8..15
    return r;
}

__device__ __forceinline__ v8f wmma_f16(v16h a, v16h b, v8f c) {
    return __builtin_amdgcn_wmma_f32_16x16x32_f16(
        /*neg_a=*/false, a, /*neg_b=*/false, b,
        /*c_mod=*/(short)0, c, /*reuse_a=*/false, /*reuse_b=*/false);
}

// ---------------------------------------------------------------------------
// 0) fp32 -> fp16 cast
// ---------------------------------------------------------------------------
__global__ void cast_f32_to_f16(const float* __restrict__ src,
                                _Float16* __restrict__ dst, int n) {
    int i = blockIdx.x * blockDim.x + threadIdx.x;
    if (i < n) dst[i] = (_Float16)src[i];
}

// ---------------------------------------------------------------------------
// 1) fused QKV projection: y = x @ W^T + b for W in {Wq,Wk,Wv} (blockIdx.z)
//    wave -> 16(M) x 64(N) output tile, K = 1024 in steps of 32.
//    q: (BH,S,64) scaled by QSCALE; k: (BH,S,64); v transposed: (BH,64,S)
// ---------------------------------------------------------------------------
__global__ void __launch_bounds__(128)
qkv_gemm_kernel(const _Float16* __restrict__ xh,
                const _Float16* __restrict__ wq,
                const _Float16* __restrict__ wk,
                const _Float16* __restrict__ wv,
                const float* __restrict__ bq,
                const float* __restrict__ bk,
                const float* __restrict__ bv,
                _Float16* __restrict__ qb,
                _Float16* __restrict__ kb,
                _Float16* __restrict__ vb) {
    const int wave = threadIdx.x >> 5;
    const int lane = threadIdx.x & 31;
    const int half = lane >> 4;
    const int l16  = lane & 15;

    const int mtile = blockIdx.x * 4 + wave;   // 0..255  (M = 4096)
    const int ngrp  = blockIdx.y;              // 0..15   (N = 1024, 64 per wave)
    const int which = blockIdx.z;              // 0=q 1=k 2=v

    const _Float16* W    = (which == 0) ? wq : (which == 1) ? wk : wv;
    const float*    bias = (which == 0) ? bq : (which == 1) ? bk : bv;

    const _Float16* Abase = xh + (size_t)(mtile * 16 + l16) * DMOD;

    v8f acc[4];
#pragma unroll
    for (int t = 0; t < 4; ++t) acc[t] = (v8f)0.0f;

    for (int kk = 0; kk < DMOD; kk += 32) {
        const _Float16* ap = Abase + kk + half * 16;
        v16h a = load16h(ap, ap + 8);
#pragma unroll
        for (int t = 0; t < 4; ++t) {
            const _Float16* bp =
                W + (size_t)(ngrp * 64 + t * 16 + l16) * DMOD + kk + half * 16;
            v16h bt = load16h(bp, bp + 8);
            acc[t] = wmma_f16(a, bt, acc[t]);
        }
    }

#pragma unroll
    for (int t = 0; t < 4; ++t) {
        const int col = ngrp * 64 + t * 16 + l16;     // n in [0,1024)
        const float bcol = bias[col];
        const int h = col >> 6, d = col & 63;
#pragma unroll
        for (int r = 0; r < 8; ++r) {
            const int row = mtile * 16 + r + 8 * half; // m in [0,4096)
            const int bb = row >> 11, s = row & (SEQ - 1);
            const int bh = bb * NHEAD + h;
            const float val = acc[t][r] + bcol;
            if (which == 0)
                qb[((size_t)bh * SEQ + s) * HDIM + d] = (_Float16)(val * QSCALE);
            else if (which == 1)
                kb[((size_t)bh * SEQ + s) * HDIM + d] = (_Float16)val;
            else
                vb[((size_t)bh * HDIM + d) * SEQ + s] = (_Float16)val;
        }
    }
}

// ---------------------------------------------------------------------------
// 2) flash attention, causal. One wave = 16 query rows of one (b,h).
//    key blocks of 32: scores 16x32 (4 wmma), online softmax, P via LDS
//    C-layout -> A-layout, then P@V (4 wmma into 16x64 f32 acc).
// ---------------------------------------------------------------------------
__global__ void __launch_bounds__(128)
attn_kernel(const _Float16* __restrict__ qb,
            const _Float16* __restrict__ kbuf,
            const _Float16* __restrict__ vb,
            _Float16* __restrict__ ob) {
    __shared__ __align__(16) _Float16 plds[4][16 * 32];

    const int wave = threadIdx.x >> 5;
    const int lane = threadIdx.x & 31;
    const int half = lane >> 4;
    const int l16  = lane & 15;

    const int bh    = blockIdx.y;                 // 0..31
    const int qtile = blockIdx.x * 4 + wave;      // 0..127
    const int q0    = qtile * 16;

    // load A = q tile (16 x 64), two K=32 steps, kept in regs for whole loop
    v16h aq[2];
#pragma unroll
    for (int kk = 0; kk < 2; ++kk) {
        const _Float16* ap =
            qb + ((size_t)bh * SEQ + q0 + l16) * HDIM + kk * 32 + half * 16;
        aq[kk] = load16h(ap, ap + 8);
    }

    v8f acc[4];
#pragma unroll
    for (int t = 0; t < 4; ++t) acc[t] = (v8f)0.0f;
    float m_i[8], l_i[8];
#pragma unroll
    for (int r = 0; r < 8; ++r) { m_i[r] = -3.0e38f; l_i[r] = 0.0f; }

    const int nblk = ((q0 + 15) >> 5) + 1;   // causal: keys up to q0+15
    for (int kbi = 0; kbi < nblk; ++kbi) {
        const int kb0 = kbi * 32;

        // ---- scores: S = q @ k^T  (16 x 32) ----
        v8f sc[2];
        sc[0] = (v8f)0.0f; sc[1] = (v8f)0.0f;
#pragma unroll
        for (int kk = 0; kk < 2; ++kk) {
#pragma unroll
            for (int t = 0; t < 2; ++t) {
                const _Float16* kp = kbuf +
                    ((size_t)bh * SEQ + kb0 + t * 16 + l16) * HDIM +
                    kk * 32 + half * 16;
                v16h bt = load16h(kp, kp + 8);
                sc[t] = wmma_f16(aq[kk], bt, sc[t]);
            }
        }

        // ---- causal mask + row max (per-element r; rows split by lane half)
        float mx[8];
        const int c0 = kb0 + l16, c1 = c0 + 16;
#pragma unroll
        for (int r = 0; r < 8; ++r) {
            const int row = q0 + r + 8 * half;
            float s0 = sc[0][r], s1 = sc[1][r];
            if (c0 > row) s0 = -3.0e38f;
            if (c1 > row) s1 = -3.0e38f;
            sc[0][r] = s0; sc[1][r] = s1;
            mx[r] = fmaxf(s0, s1);
        }
#pragma unroll
        for (int off = 1; off < 16; off <<= 1)
#pragma unroll
            for (int r = 0; r < 8; ++r)
                mx[r] = fmaxf(mx[r], __shfl_xor(mx[r], off, 32));

        // ---- online softmax update ----
        float alpha[8], rs[8];
#pragma unroll
        for (int r = 0; r < 8; ++r) {
            const float mn = fmaxf(m_i[r], mx[r]);
            alpha[r] = exp2f(m_i[r] - mn);
            m_i[r] = mn;
        }
#pragma unroll
        for (int r = 0; r < 8; ++r) {
            const float p0 = exp2f(sc[0][r] - m_i[r]);
            const float p1 = exp2f(sc[1][r] - m_i[r]);
            const int rl = r + 8 * half;
            plds[wave][rl * 32 + l16]      = (_Float16)p0;
            plds[wave][rl * 32 + 16 + l16] = (_Float16)p1;
            rs[r] = p0 + p1;
        }
#pragma unroll
        for (int off = 1; off < 16; off <<= 1)
#pragma unroll
            for (int r = 0; r < 8; ++r)
                rs[r] += __shfl_xor(rs[r], off, 32);
#pragma unroll
        for (int r = 0; r < 8; ++r) l_i[r] = l_i[r] * alpha[r] + rs[r];
#pragma unroll
        for (int t = 0; t < 4; ++t)
#pragma unroll
            for (int r = 0; r < 8; ++r) acc[t][r] *= alpha[r];

        // intra-wave LDS store->load (C-layout -> A-layout for P)
        asm volatile("s_wait_dscnt 0" ::: "memory");
        const _Float16* pp = &plds[wave][l16 * 32 + half * 16];
        v16h pa = load16h(pp, pp + 8);

        // ---- acc += P @ V  (V transposed: (dh, s) row-major = (N,K)) ----
#pragma unroll
        for (int t = 0; t < 4; ++t) {
            const _Float16* vp = vb +
                ((size_t)bh * HDIM + t * 16 + l16) * SEQ + kb0 + half * 16;
            v16h bt = load16h(vp, vp + 8);
            acc[t] = wmma_f16(pa, bt, acc[t]);
        }
    }

    // ---- epilogue: normalize, write (b, s, h*64+d) f16 row-major ----
    const int b = bh >> 4, h = bh & (NHEAD - 1);
#pragma unroll
    for (int r = 0; r < 8; ++r) {
        const float rl = 1.0f / l_i[r];
        const int s = q0 + r + 8 * half;
        const size_t mrow = (size_t)(b * SEQ + s) * DMOD + h * HDIM;
#pragma unroll
        for (int t = 0; t < 4; ++t)
            ob[mrow + t * 16 + l16] = (_Float16)(acc[t][r] * rl);
    }
}

// ---------------------------------------------------------------------------
// 3) output projection: out = attn @ Wo^T + bo  (fp32 out)
// ---------------------------------------------------------------------------
__global__ void __launch_bounds__(128)
o_gemm_kernel(const _Float16* __restrict__ ah,
              const _Float16* __restrict__ wo,
              const float* __restrict__ bo,
              float* __restrict__ out) {
    const int wave = threadIdx.x >> 5;
    const int lane = threadIdx.x & 31;
    const int half = lane >> 4;
    const int l16  = lane & 15;

    const int mtile = blockIdx.x * 4 + wave;
    const int ngrp  = blockIdx.y;

    const _Float16* Abase = ah + (size_t)(mtile * 16 + l16) * DMOD;

    v8f acc[4];
#pragma unroll
    for (int t = 0; t < 4; ++t) acc[t] = (v8f)0.0f;

    for (int kk = 0; kk < DMOD; kk += 32) {
        const _Float16* ap = Abase + kk + half * 16;
        v16h a = load16h(ap, ap + 8);
#pragma unroll
        for (int t = 0; t < 4; ++t) {
            const _Float16* bp =
                wo + (size_t)(ngrp * 64 + t * 16 + l16) * DMOD + kk + half * 16;
            v16h bt = load16h(bp, bp + 8);
            acc[t] = wmma_f16(a, bt, acc[t]);
        }
    }

#pragma unroll
    for (int t = 0; t < 4; ++t) {
        const int col = ngrp * 64 + t * 16 + l16;
        const float bcol = bo[col];
#pragma unroll
        for (int r = 0; r < 8; ++r) {
            const int row = mtile * 16 + r + 8 * half;
            out[(size_t)row * DMOD + col] = acc[t][r] + bcol;
        }
    }
}

// ---------------------------------------------------------------------------
// launcher
// ---------------------------------------------------------------------------
extern "C" void kernel_launch(void* const* d_in, const int* in_sizes, int n_in,
                              void* d_out, int out_size, void* d_ws, size_t ws_size,
                              hipStream_t stream) {
    const float* x  = (const float*)d_in[0];
    // d_in[1] = attn_mask (causal tril) -> implemented analytically
    const float* Wq = (const float*)d_in[2];
    const float* bq = (const float*)d_in[3];
    const float* Wk = (const float*)d_in[4];
    const float* bk = (const float*)d_in[5];
    const float* Wv = (const float*)d_in[6];
    const float* bv = (const float*)d_in[7];
    const float* Wo = (const float*)d_in[8];
    const float* bo = (const float*)d_in[9];
    float* out = (float*)d_out;

    const size_t NX = (size_t)BATCH * SEQ * DMOD;   // 4194304
    const size_t NW = (size_t)DMOD * DMOD;          // 1048576

    _Float16* ws  = (_Float16*)d_ws;
    _Float16* xh  = ws;              size_t off = NX;
    _Float16* wqh = ws + off;        off += NW;
    _Float16* wkh = ws + off;        off += NW;
    _Float16* wvh = ws + off;        off += NW;
    _Float16* woh = ws + off;        off += NW;
    _Float16* qb  = ws + off;        off += NX;     // (BH, S, 64)
    _Float16* kb  = ws + off;        off += NX;     // (BH, S, 64)
    _Float16* vb  = ws + off;        off += NX;     // (BH, 64, S)
    _Float16* ob  = ws + off;        off += NX;     // (B*S, D)

    cast_f32_to_f16<<<(int)((NX + 255) / 256), 256, 0, stream>>>(x,  xh,  (int)NX);
    cast_f32_to_f16<<<(int)((NW + 255) / 256), 256, 0, stream>>>(Wq, wqh, (int)NW);
    cast_f32_to_f16<<<(int)((NW + 255) / 256), 256, 0, stream>>>(Wk, wkh, (int)NW);
    cast_f32_to_f16<<<(int)((NW + 255) / 256), 256, 0, stream>>>(Wv, wvh, (int)NW);
    cast_f32_to_f16<<<(int)((NW + 255) / 256), 256, 0, stream>>>(Wo, woh, (int)NW);

    qkv_gemm_kernel<<<dim3(64, 16, 3), 128, 0, stream>>>(
        xh, wqh, wkh, wvh, bq, bk, bv, qb, kb, vb);

    attn_kernel<<<dim3(32, BATCH * NHEAD), 128, 0, stream>>>(qb, kb, vb, ob);

    o_gemm_kernel<<<dim3(64, 16), 128, 0, stream>>>(ob, woh, bo, out);
}
